// Social_Aggregator_21148418965783
// MI455X (gfx1250) — matrix operand order repaired and checked by reading, not compiled, not run on hardware
//
#include <hip/hip_runtime.h>
#include <hip/hip_bf16.h>

typedef __attribute__((ext_vector_type(16))) _Float16 v16h;
typedef __attribute__((ext_vector_type(8)))  _Float16 v8h;
typedef __attribute__((ext_vector_type(8)))  float    v8f;

#define WAVES_PER_BLOCK 4

// Load one 16x32 f16 A-operand row-slice for this lane:
// lane<16 : K = {kb..kb+7, kb+16..kb+23} with kb = base
// lane>=16: same but base already shifted by +8 by the caller.
__device__ __forceinline__ v16h ld_a(const _Float16* base) {
    v8h lo = *(const v8h*)(base);
    v8h hi = *(const v8h*)(base + 16);
    return __builtin_shufflevector(lo, hi, 0, 1, 2, 3, 4, 5, 6, 7,
                                           8, 9, 10, 11, 12, 13, 14, 15);
}

__global__ __launch_bounds__(128) void graphrec_att_kernel(
    const int* __restrict__ nodes,
    const int* __restrict__ neighs,
    const unsigned char* __restrict__ mask,
    const float* __restrict__ u2e,
    const float* __restrict__ W1,
    const float* __restrict__ b1,
    const float* __restrict__ W2,
    const float* __restrict__ b2,
    const float* __restrict__ W3,
    const float* __restrict__ b3,
    float* __restrict__ out,
    int N)
{
    // B operands pre-swizzled: [kt][nt][lane][16 halfs contiguous]
    __shared__ __align__(32) _Float16 sW1b[4 * 4 * 32 * 16];   // 16 KB
    __shared__ __align__(32) _Float16 sW2b[2 * 4 * 32 * 16];   //  8 KB
    __shared__ float sB1[64];
    __shared__ float sB2[64];
    __shared__ float sW3[64];
    __shared__ float sB3;
    // Per-wave x = [e_u (cols 0..63) | u_rep (cols 64..127)] f16.
    // Cols 64..127 are recycled as the hidden activation h after layer 1.
    __shared__ __align__(16) _Float16 sX[WAVES_PER_BLOCK][32 * 128]; // 32 KB
    __shared__ float sS[WAVES_PER_BLOCK][32];

    const int tid = threadIdx.x;

    // ---- block setup: swizzle weights into WMMA B-operand layout ----
    for (int f = tid; f < 4 * 4 * 32 * 16; f += 128) {
        const int e  = f & 15;
        const int ln = (f >> 4) & 31;
        const int nt = (f >> 9) & 3;
        const int kt = f >> 11;
        const int krow = kt * 32 + ((ln & 16) ? 8 : 0) + (e & 7) + ((e & 8) ? 16 : 0);  // K of B
        const int ncol = nt * 16 + (ln & 15);                 // N of B
        sW1b[f] = (_Float16)W1[krow * 64 + ncol];
    }
    for (int f = tid; f < 2 * 4 * 32 * 16; f += 128) {
        const int e  = f & 15;
        const int ln = (f >> 4) & 31;
        const int nt = (f >> 9) & 3;
        const int kt = f >> 11;
        const int krow = kt * 32 + ((ln & 16) ? 8 : 0) + (e & 7) + ((e & 8) ? 16 : 0);
        const int ncol = nt * 16 + (ln & 15);
        sW2b[f] = (_Float16)W2[krow * 64 + ncol];
    }
    if (tid < 64) { sB1[tid] = b1[tid]; sB2[tid] = b2[tid]; sW3[tid] = W3[tid]; }
    if (tid == 0) sB3 = b3[0];

    const int wave = tid >> 5;
    const int lane = tid & 31;
    const int n = blockIdx.x * WAVES_PER_BLOCK + wave;
    const bool valid = (n < N);
    const int nc = valid ? n : 0;   // clamp: keep EXEC full for WMMA

    _Float16* X = &sX[wave][0];

    // ---- fill x tile: lane owns neighbor-row k == lane ----
    {
        const long nb = (long)neighs[(long)nc * 32 + lane];
        const float4* np = (const float4*)(u2e + nb * 64);
        const float4* up = (const float4*)(u2e + (long)nodes[nc] * 64);
        _Float16* xr = X + lane * 128;
#pragma unroll
        for (int c4 = 0; c4 < 16; ++c4) {
            const float4 v = np[c4];
            xr[c4 * 4 + 0] = (_Float16)v.x;
            xr[c4 * 4 + 1] = (_Float16)v.y;
            xr[c4 * 4 + 2] = (_Float16)v.z;
            xr[c4 * 4 + 3] = (_Float16)v.w;
            const float4 u = up[c4];
            xr[64 + c4 * 4 + 0] = (_Float16)u.x;
            xr[64 + c4 * 4 + 1] = (_Float16)u.y;
            xr[64 + c4 * 4 + 2] = (_Float16)u.z;
            xr[64 + c4 * 4 + 3] = (_Float16)u.w;
        }
    }
    __syncthreads();

    const int row0 = (lane & 15) * 128;
    const int vo   = (lane & 16) ? 8 : 0;   // C-layout row offset for hi half

    // ---- layer 1: [32x128] x [128x64], 32x v_wmma_f32_16x16x32_f16 ----
    v8f acc[2][4] = {};
#pragma unroll
    for (int kt = 0; kt < 4; ++kt) {
        const int kb = kt * 32 + ((lane & 16) ? 8 : 0);
        const v16h a0 = ld_a(X + row0 + kb);
        const v16h a1 = ld_a(X + row0 + 16 * 128 + kb);
#pragma unroll
        for (int nt = 0; nt < 4; ++nt) {
            const v16h b = *(const v16h*)&sW1b[((kt * 4 + nt) * 32 + lane) * 16];
            acc[0][nt] = __builtin_amdgcn_wmma_f32_16x16x32_f16(
                false, a0, false, b, (short)0, acc[0][nt], false, false);
            acc[1][nt] = __builtin_amdgcn_wmma_f32_16x16x32_f16(
                false, a1, false, b, (short)0, acc[1][nt], false, false);
        }
    }
    __syncthreads();

    // ---- bias + relu, write h into cols 64..127 of X ----
#pragma unroll
    for (int mt = 0; mt < 2; ++mt) {
#pragma unroll
        for (int nt = 0; nt < 4; ++nt) {
            const int ncol = nt * 16 + (lane & 15);
            const float bias = sB1[ncol];
#pragma unroll
            for (int v = 0; v < 8; ++v) {
                float h = acc[mt][nt][v] + bias;
                h = h > 0.0f ? h : 0.0f;
                X[(mt * 16 + vo + v) * 128 + 64 + ncol] = (_Float16)h;
            }
        }
    }
    __syncthreads();

    // ---- layer 2: [32x64] x [64x64], 16x WMMA ----
    v8f acc2[2][4] = {};
#pragma unroll
    for (int kt = 0; kt < 2; ++kt) {
        const int kb = 64 + kt * 32 + ((lane & 16) ? 8 : 0);
        const v16h a0 = ld_a(X + row0 + kb);
        const v16h a1 = ld_a(X + row0 + 16 * 128 + kb);
#pragma unroll
        for (int nt = 0; nt < 4; ++nt) {
            const v16h b = *(const v16h*)&sW2b[((kt * 4 + nt) * 32 + lane) * 16];
            acc2[0][nt] = __builtin_amdgcn_wmma_f32_16x16x32_f16(
                false, a0, false, b, (short)0, acc2[0][nt], false, false);
            acc2[1][nt] = __builtin_amdgcn_wmma_f32_16x16x32_f16(
                false, a1, false, b, (short)0, acc2[1][nt], false, false);
        }
    }

    // ---- bias + relu + dot with W3 -> per-neighbor score ----
    float ps[2][8] = {};
#pragma unroll
    for (int mt = 0; mt < 2; ++mt) {
#pragma unroll
        for (int nt = 0; nt < 4; ++nt) {
            const int ncol = nt * 16 + (lane & 15);
            const float bias = sB2[ncol];
            const float w3v = sW3[ncol];
#pragma unroll
            for (int v = 0; v < 8; ++v) {
                float h = acc2[mt][nt][v] + bias;
                h = h > 0.0f ? h : 0.0f;
                ps[mt][v] += h * w3v;
            }
        }
    }
    // reduce partial dot over the 16 lanes of each half (each lane holds one N col set)
#pragma unroll
    for (int mt = 0; mt < 2; ++mt) {
#pragma unroll
        for (int v = 0; v < 8; ++v) {
#pragma unroll
            for (int m = 1; m < 16; m <<= 1)
                ps[mt][v] += __shfl_xor(ps[mt][v], m, 32);
        }
    }
    if (lane == 0 || lane == 16) {
#pragma unroll
        for (int mt = 0; mt < 2; ++mt)
#pragma unroll
            for (int v = 0; v < 8; ++v)
                sS[wave][mt * 16 + vo + v] = ps[mt][v] + sB3;
    }
    __syncthreads();

    // ---- masked softmax over K=32 (lane k holds score k) ----
    float s = sS[wave][lane];
    if (!mask[(long)nc * 32 + lane]) s = -1e9f;
    float mx = s;
#pragma unroll
    for (int off = 1; off < 32; off <<= 1)
        mx = fmaxf(mx, __shfl_xor(mx, off, 32));
    const float ev = __expf(s - mx);
    float sum = ev;
#pragma unroll
    for (int off = 1; off < 32; off <<= 1)
        sum += __shfl_xor(sum, off, 32);
    const float att = ev / sum;

    // ---- att-weighted aggregation of e_u (f16 cols 0..63 of X) ----
    float ox = 0.0f, oy = 0.0f;
    const int c0 = lane * 2;
#pragma unroll 4
    for (int k = 0; k < 32; ++k) {
        const float a = __shfl(att, k, 32);
        ox += a * (float)X[k * 128 + c0];
        oy += a * (float)X[k * 128 + c0 + 1];
    }
    if (valid) {
        out[(long)n * 64 + c0]     = ox;
        out[(long)n * 64 + c0 + 1] = oy;
    }
}

extern "C" void kernel_launch(void* const* d_in, const int* in_sizes, int n_in,
                              void* d_out, int out_size, void* d_ws, size_t ws_size,
                              hipStream_t stream) {
    (void)n_in; (void)out_size; (void)d_ws; (void)ws_size;
    const int* nodes  = (const int*)d_in[0];
    const int* neighs = (const int*)d_in[1];
    const unsigned char* mask = (const unsigned char*)d_in[2];
    const float* u2e = (const float*)d_in[3];
    const float* W1  = (const float*)d_in[4];
    const float* b1  = (const float*)d_in[5];
    const float* W2  = (const float*)d_in[6];
    const float* b2  = (const float*)d_in[7];
    const float* W3  = (const float*)d_in[8];
    const float* b3  = (const float*)d_in[9];
    float* out = (float*)d_out;

    const int N = in_sizes[0];
    const int blocks = (N + WAVES_PER_BLOCK - 1) / WAVES_PER_BLOCK;
    graphrec_att_kernel<<<blocks, 128, 0, stream>>>(
        nodes, neighs, mask, u2e, W1, b1, W2, b2, W3, b3, out, N);
}
